// Seq2Seq_65627100283641
// MI455X (gfx1250) — matrix operand order repaired and checked by reading, not compiled, not run on hardware
//
#include <hip/hip_runtime.h>

// ---------------------------------------------------------------------------
// Seq2Seq LSTM (enc 512 + dec 256 steps), B=2048, H=50 padded to K=64,
// gates 4H=200 padded to 256. One wave32 owns 16 batch rows persistently.
//
// Gate GEMM per step: D[16x256] = A[16x64] x B[64x256], 16 N-tiles x 2 K-frags
// = 32x v_wmma_f32_16x16x32_bf16.  Input term and bias are FOLDED INTO K:
//   A[m][0..49] = h,  A[m][50] = x_m,  A[m][51] = 1.0
//   B[0..49][n] = W_hh^T, B[50][n] = W_ih[n], B[51][n] = b_ih[n]+b_hh[n]
// so accumulators start at inline zero. The compiler hoists the loop-invariant
// B fragments from LDS into registers: steady state is 4x ds_load_b128 (A) +
// 32 WMMA + activations per step. Activations use raw v_exp_f32/v_rcp_f32
// (no IEEE division expansion).
// ---------------------------------------------------------------------------

typedef __attribute__((ext_vector_type(16))) __bf16 v16bf;
typedef __attribute__((ext_vector_type(8)))  float  v8f;

#define HID     50
#define SEQ     512
#define BATCH   2048
#define TLEN    256
#define KPAD    64          // padded K: 50 hidden + x@50 + 1.0@51 + zeros
#define NTILES  16          // 256 padded gate columns / 16
#define WAVES   2           // waves per block
#define NTHREADS (WAVES * 32)

#define LOG2E 1.44269504088896340736f

union FragU { unsigned int u[8]; v16bf v; };

__device__ __forceinline__ unsigned int f2bf(float f) {
    unsigned int u = __float_as_uint(f);
    return (u + 0x7FFFu + ((u >> 16) & 1u)) >> 16;   // round-to-nearest-even
}
// sigmoid(x) = 1/(1+2^(-x*log2e)) using raw trans ops (v_exp_f32, v_rcp_f32)
__device__ __forceinline__ float sigf(float x) {
    return __builtin_amdgcn_rcpf(1.0f + __builtin_amdgcn_exp2f(-LOG2E * x));
}
// tanh(x) = 1 - 2/(exp(2x)+1)
__device__ __forceinline__ float tanh_f(float x) {
    float e = __builtin_amdgcn_exp2f((2.0f * LOG2E) * x);
    return 1.0f - 2.0f * __builtin_amdgcn_rcpf(e + 1.0f);
}

// LDS weight image, WMMA B-fragment order:
//   s_wB[((ntile*2+frag)*8 + j)*32 + lane] = packed {w[k+1][n], w[k][n]} bf16
//   n = ntile*16 + lane%16 (gate col, 64-padded per gate)
//   k = frag*32 + 16*(lane/16) + 2j
// with augmented rows k=50 (W_ih) and k=51 (combined bias).
__device__ __forceinline__ void fill_weights(const float* __restrict__ Whh,
                                             const float* __restrict__ Wih,
                                             const float* __restrict__ bih,
                                             const float* __restrict__ bhh,
                                             unsigned int* s_wB, int tid) {
    for (int idx = tid; idx < NTILES * 2 * 8 * 32; idx += NTHREADS) {
        int ln = idx & 31, j = (idx >> 5) & 7, frag = (idx >> 8) & 1, nt = idx >> 9;
        int n = nt * 16 + (ln & 15);
        int gate = n >> 6, hho = n & 63;                 // gate-output row (64-padded)
        int k = frag * 32 + ((ln >> 4) << 4) + (j << 1); // augmented K index (even)
        float w0 = 0.f, w1 = 0.f;
        if (hho < HID) {
            int r = gate * HID + hho;
            if (k < HID) {                               // recurrent weights
                const float* row = Whh + r * HID;        // W_hh [200][50] row-major
                w0 = row[k];
                if (k + 1 < HID) w1 = row[k + 1];
            } else if (k == HID) {                       // k=50: W_ih, k=51: bias
                w0 = Wih[r];                             // W_ih is [4H][1]
                w1 = bih[r] + bhh[r];
            }
        }
        s_wB[idx] = f2bf(w0) | (f2bf(w1) << 16);
    }
}

// One LSTM step for this wave's 16 batch rows. Caller has already written the
// augmented A entries {x_m, 1.0} at k=50..51 of each row in hrow.
template <bool DEC>
__device__ __forceinline__ void lstm_step(const unsigned int* __restrict__ s_wB,
                                          unsigned short* hrow, int lane,
                                          v8f* cst, float* ypart, const float* fcw) {
    const int lhi = lane >> 4, nloc = lane & 15;
    // --- A fragments (h | x | 1 | 0-pad), 16x64 bf16, from LDS ---
    FragU a0, a1;
    const int m0 = lane & 15;
#pragma unroll
    for (int j = 0; j < 8; ++j) {
        int k0 = (lhi << 4) + (j << 1);
        a0.u[j] = *(const unsigned int*)(hrow + m0 * KPAD + k0);
        a1.u[j] = *(const unsigned int*)(hrow + m0 * KPAD + 32 + k0);
    }
    const v8f zero = {};
#pragma unroll
    for (int t = 0; t < 4; ++t) {        // 4 hidden-column tiles per gate
        v8f acc[4];
#pragma unroll
        for (int g = 0; g < 4; ++g) {    // i, f, g, o
            const int nt = g * 4 + t;
            FragU b0, b1;
            const unsigned int* wp = s_wB + nt * 512;
#pragma unroll
            for (int j = 0; j < 8; ++j) {
                b0.u[j] = wp[j * 32 + lane];
                b1.u[j] = wp[256 + j * 32 + lane];
            }
            acc[g] = __builtin_amdgcn_wmma_f32_16x16x32_bf16(
                false, a0.v, false, b0.v, (short)0, zero, false, false);
            acc[g] = __builtin_amdgcn_wmma_f32_16x16x32_bf16(
                false, a1.v, false, b1.v, (short)0, acc[g], false, false);
        }
        // --- activations + state update (D layout: lane=col, vgpr=row) ---
#pragma unroll
        for (int v = 0; v < 8; ++v) {
            float ig = sigf(acc[0][v]);
            float fg = sigf(acc[1][v]);
            float gg = tanh_f(acc[2][v]);
            float og = sigf(acc[3][v]);
            float c  = fg * cst[t][v] + ig * gg;
            cst[t][v] = c;
            float hv = og * tanh_f(c);
            int m = (lhi << 3) + v;
            hrow[m * KPAD + t * 16 + nloc] = (unsigned short)f2bf(hv);
            if (DEC) ypart[v] += hv * fcw[t];
        }
    }
}

__global__ __launch_bounds__(NTHREADS)
void seq2seq_kernel(const float* __restrict__ src,
                    const float* __restrict__ enc_Wih, const float* __restrict__ enc_Whh,
                    const float* __restrict__ enc_bih, const float* __restrict__ enc_bhh,
                    const float* __restrict__ dec_Wih, const float* __restrict__ dec_Whh,
                    const float* __restrict__ dec_bih, const float* __restrict__ dec_bhh,
                    const float* __restrict__ fcW, const float* __restrict__ fcb_p,
                    float* __restrict__ out) {
    __shared__ unsigned int   s_wB[NTILES * 2 * 8 * 32];   // 32 KB swizzled weights
    __shared__ unsigned short s_h[WAVES][16 * KPAD];       // per-wave augmented A (bf16)

    const int tid   = threadIdx.x;
    const int lane  = tid & 31;
    const int wave  = tid >> 5;
    const int lhi   = lane >> 4;
    const int nloc  = lane & 15;
    const int mbase = (blockIdx.x * WAVES + wave) * 16;
    unsigned short* hrow = s_h[wave];

    // h0 = 0 (wave-private buffer, no barrier needed)
    for (int i = lane; i < 16 * KPAD; i += 32) hrow[i] = 0;

    v8f cst[4];
#pragma unroll
    for (int t = 0; t < 4; ++t)
#pragma unroll
        for (int v = 0; v < 8; ++v) cst[t][v] = 0.f;

    // ===================== encoder =====================
    fill_weights(enc_Whh, enc_Wih, enc_bih, enc_bhh, s_wB, tid);
    __syncthreads();

    for (int t = 0; t < SEQ; ++t) {
        // augmented A entries: {bf16(x_m), bf16(1.0)} packed at k=50..51
        if (nloc == 0) {
#pragma unroll
            for (int v = 0; v < 8; ++v) {
                int m = (lhi << 3) + v;
                const float* p = &src[(mbase + m) * SEQ + t];
                float x = *p;
                __builtin_prefetch(p + 4, 0, 0);   // global_prefetch_b8
                *(unsigned int*)(hrow + m * KPAD + HID) = f2bf(x) | 0x3F800000u;
            }
        }
        lstm_step<false>(s_wB, hrow, lane, cst, nullptr, nullptr);
    }

    // ===================== decoder =====================
    __syncthreads();   // everyone done reading encoder weights
    fill_weights(dec_Whh, dec_Wih, dec_bih, dec_bhh, s_wB, tid);
    __syncthreads();

    float fcw[4];
#pragma unroll
    for (int t = 0; t < 4; ++t) {
        int hh = t * 16 + nloc;
        fcw[t] = (hh < HID) ? fcW[hh] : 0.f;
    }
    const float fcb = fcb_p[0];

    float xv[8];
#pragma unroll
    for (int v = 0; v < 8; ++v) xv[v] = 0.f;       // decoder x0 = zeros

    for (int s = 0; s < TLEN; ++s) {
        if (nloc == 0) {
#pragma unroll
            for (int v = 0; v < 8; ++v) {
                int m = (lhi << 3) + v;
                *(unsigned int*)(hrow + m * KPAD + HID) = f2bf(xv[v]) | 0x3F800000u;
            }
        }
        float yp[8];
#pragma unroll
        for (int v = 0; v < 8; ++v) yp[v] = 0.f;
        lstm_step<true>(s_wB, hrow, lane, cst, yp, fcw);
        // y[m] = sum over hidden (spread across the 16 lanes of each half)
#pragma unroll
        for (int v = 0; v < 8; ++v) {
            float s0 = yp[v];
            s0 += __shfl_xor(s0, 1, 32);
            s0 += __shfl_xor(s0, 2, 32);
            s0 += __shfl_xor(s0, 4, 32);
            s0 += __shfl_xor(s0, 8, 32);           // stays within 16-lane half
            float y = s0 + fcb;
            if (nloc == 0)
                out[(mbase + (lhi << 3) + v) * TLEN + s] = y;
            xv[v] = y;                              // feedback: next decoder input
        }
    }
}

extern "C" void kernel_launch(void* const* d_in, const int* in_sizes, int n_in,
                              void* d_out, int out_size, void* d_ws, size_t ws_size,
                              hipStream_t stream) {
    (void)in_sizes; (void)n_in; (void)d_ws; (void)ws_size; (void)out_size;
    const float* src     = (const float*)d_in[0];
    const float* enc_Wih = (const float*)d_in[1];
    const float* enc_Whh = (const float*)d_in[2];
    const float* enc_bih = (const float*)d_in[3];
    const float* enc_bhh = (const float*)d_in[4];
    const float* dec_Wih = (const float*)d_in[5];
    const float* dec_Whh = (const float*)d_in[6];
    const float* dec_bih = (const float*)d_in[7];
    const float* dec_bhh = (const float*)d_in[8];
    const float* fcW     = (const float*)d_in[9];
    const float* fcb     = (const float*)d_in[10];
    // d_in[11] = target_len (hardcoded TLEN=256 to match reference)

    dim3 grid(BATCH / (WAVES * 16));   // 64 blocks x 2 waves x 16 rows = 2048
    dim3 block(NTHREADS);
    hipLaunchKernelGGL(seq2seq_kernel, grid, block, 0, stream,
                       src, enc_Wih, enc_Whh, enc_bih, enc_bhh,
                       dec_Wih, dec_Whh, dec_bih, dec_bhh, fcW, fcb,
                       (float*)d_out);
}
